// T5EncoderSelfAttention_13451837571868
// MI455X (gfx1250) — compile-verified, matrix operand order checked by reading, not run on hardware
//
#include <hip/hip_runtime.h>
#include <hip/hip_bf16.h>
#include <stdint.h>

// ---------------------------------------------------------------------------
// Problem constants (T5 encoder self-attention)
// ---------------------------------------------------------------------------
#define BB   2
#define SS   2048
#define DD   1024
#define HH   16
#define HDD  64
#define MM   (BB * SS)      // 4096 rows
#define NQKV (3 * HH * HDD) // 3072
#define NEGV (-10000.0f)

typedef __attribute__((ext_vector_type(16))) __bf16 v16bf;
typedef __attribute__((ext_vector_type(8)))  float  v8f;

// float -> bf16 round-to-nearest-even via bit ops
static __device__ __forceinline__ __bf16 f2bf(float f) {
    unsigned u = __float_as_uint(f);
    u += 0x7FFFu + ((u >> 16) & 1u);
    unsigned short h = (unsigned short)(u >> 16);
    __bf16 r;
    __builtin_memcpy(&r, &h, sizeof(r));
    return r;
}

// WMMA 16-bit A/B fragment load (ISA 7.12.2 pattern): two 16B chunks from the
// lane's row (row = lane&15) at k-offsets kbase+(lane>>4)*8 and +16.
static __device__ __forceinline__ v16bf load_frag(const __bf16* rowbase, int kbase, int lane) {
    int kb = kbase + ((lane >> 4) << 3);
    union { v16bf v; uint4 q[2]; } u;
    u.q[0] = *(const uint4*)(rowbase + kb);
    u.q[1] = *(const uint4*)(rowbase + kb + 16);
    return u.v;
}

static __device__ __forceinline__ float redmax16(float x) {
    x = fmaxf(x, __shfl_xor(x, 1, 32));
    x = fmaxf(x, __shfl_xor(x, 2, 32));
    x = fmaxf(x, __shfl_xor(x, 4, 32));
    x = fmaxf(x, __shfl_xor(x, 8, 32));
    return x;
}
static __device__ __forceinline__ float redsum16(float x) {
    x += __shfl_xor(x, 1, 32);
    x += __shfl_xor(x, 2, 32);
    x += __shfl_xor(x, 4, 32);
    x += __shfl_xor(x, 8, 32);
    return x;
}

// ---- CDNA5 async global->LDS copy (ASYNCcnt path) -------------------------
static __device__ __forceinline__ uint32_t lds_off(const void* p) {
    return (uint32_t)(uintptr_t)p;   // low 32 bits of generic ptr = LDS offset
}
static __device__ __forceinline__ void async_b128(uint32_t ldsoff, const void* g) {
    asm volatile("global_load_async_to_lds_b128 %0, %1, off"
                 :: "v"(ldsoff), "v"(g) : "memory");
}
template <int N>
static __device__ __forceinline__ void wait_async() {
    asm volatile("s_wait_asynccnt %0" :: "i"(N) : "memory");
}

// ---------------------------------------------------------------------------
// fp32 -> bf16 conversion (4 elements / thread)
// ---------------------------------------------------------------------------
__global__ void cvt_f32_bf16(const float* __restrict__ src, __bf16* __restrict__ dst, int n4) {
    int i = blockIdx.x * blockDim.x + threadIdx.x;
    if (i < n4) {
        float4 f = ((const float4*)src)[i];
        union { __bf16 h[4]; uint2 u; } o;
        o.h[0] = f2bf(f.x); o.h[1] = f2bf(f.y);
        o.h[2] = f2bf(f.z); o.h[3] = f2bf(f.w);
        ((uint2*)dst)[i] = o.u;
    }
}

// ---------------------------------------------------------------------------
// bf16 WMMA GEMM: C[M,N] = A[M,K] * B[N,K]^T.
// Block: 256 thr (8 waves) -> 16 rows x 512 cols. A slice [16 x 128] is
// async-staged into LDS (double buffered) and shared by all 8 waves; each
// wave owns 4 N-tiles so each A fragment feeds 4 WMMAs. B fragments are
// batch-preloaded so the compiler can issue one load clause + partial waits.
// mode 0: fp32 store; mode 1: QKV scatter -> Q[b,h,s,d], K[b,h,s,d], V^T[b,h,d,s]
// ---------------------------------------------------------------------------
__global__ __launch_bounds__(256) void gemm_bf16_wmma(
    const __bf16* __restrict__ A, const __bf16* __restrict__ Bm,
    int M, int N, int K,
    float* __restrict__ outF,
    __bf16* __restrict__ Qb, __bf16* __restrict__ Kb, __bf16* __restrict__ VTb,
    int mode)
{
    __shared__ __attribute__((aligned(16))) __bf16 Atile[2][16][128];

    const int tid  = threadIdx.x;
    const int lane = tid & 31;
    const int w    = tid >> 5;
    const int mt   = blockIdx.y;
    const int nb   = blockIdx.x * 512 + w * 64;

    const __bf16* Brow[4];
#pragma unroll
    for (int t = 0; t < 4; ++t)
        Brow[t] = Bm + (size_t)(nb + t * 16 + (lane & 15)) * K;

    // A staging: 256 threads x 16B = 4KB = [16 rows][128 k] bf16
    const int arow = tid >> 4;
    const int aseg = tid & 15;
    const __bf16* Ag = A + (size_t)(mt * 16 + arow) * K + aseg * 8;

    v8f acc[4];
#pragma unroll
    for (int t = 0; t < 4; ++t) acc[t] = (v8f){};

    async_b128(lds_off(&Atile[0][arow][aseg * 8]), Ag);           // chunk 0

    const int nch = K / 128;
    for (int c = 0; c < nch; ++c) {
        const int cur = c & 1;
        if (c + 1 < nch) {
            async_b128(lds_off(&Atile[cur ^ 1][arow][aseg * 8]), Ag + (size_t)(c + 1) * 128);
            wait_async<1>();   // current buffer's load done (in-order completion)
        } else {
            wait_async<0>();
        }
        __syncthreads();
#pragma unroll
        for (int ks = 0; ks < 4; ++ks) {
            // batch-preload: 1 LDS A fragment + 4 global B fragments
            v16bf af = load_frag(&Atile[cur][lane & 15][0], ks * 32, lane);
            v16bf bf[4];
#pragma unroll
            for (int t = 0; t < 4; ++t)
                bf[t] = load_frag(Brow[t], c * 128 + ks * 32, lane);
#pragma unroll
            for (int t = 0; t < 4; ++t)
                acc[t] = __builtin_amdgcn_wmma_f32_16x16x32_bf16(
                    false, af, false, bf[t], (short)0, acc[t], false, false);
        }
        __syncthreads();
    }

    const int mbase = mt * 16 + ((lane >> 4) << 3);
    if (mode == 0) {
#pragma unroll
        for (int t = 0; t < 4; ++t) {
            const int n = nb + t * 16 + (lane & 15);
#pragma unroll
            for (int r = 0; r < 8; ++r)
                outF[(size_t)(mbase + r) * N + n] = acc[t][r];
        }
    } else {
#pragma unroll
        for (int t = 0; t < 4; ++t) {
            const int n = nb + t * 16 + (lane & 15);
            const int i = n >> 10, rem = n & 1023, h = rem >> 6, d = rem & 63;
#pragma unroll
            for (int r = 0; r < 8; ++r) {
                const int m = mbase + r;
                const int b = m >> 11, s = m & (SS - 1);
                const size_t bh = (size_t)b * HH + h;
                const __bf16 v = f2bf(acc[t][r]);
                if (i == 0)      Qb[(bh * SS + s) * HDD + d] = v;
                else if (i == 1) Kb[(bh * SS + s) * HDD + d] = v;
                else             VTb[(bh * HDD + d) * SS + s] = v;
            }
        }
    }
}

// ---------------------------------------------------------------------------
// Flash attention. Block = 128 thr (4 waves) = 4 consecutive 16-query tiles of
// the SAME (b,h): K tile [32x64] and V^T tile [64x32] per 32-key block are
// async-staged into LDS (double buffered) and shared by all 4 waves. Online
// softmax with shfl-xor row reductions; P re-laid out through a per-wave LDS
// tile (same-wave DS ops are in-order; wave_barrier = scheduling fence only).
// ---------------------------------------------------------------------------
__global__ __launch_bounds__(128) void attn_wmma(
    const __bf16* __restrict__ Qb, const __bf16* __restrict__ Kb,
    const __bf16* __restrict__ VTb,
    const float* __restrict__ bias, const unsigned char* __restrict__ maskp,
    __bf16* __restrict__ attnb)
{
    __shared__ __attribute__((aligned(16))) __bf16 Kt[2][32][64];  // keys x d
    __shared__ __attribute__((aligned(16))) __bf16 Vt[2][64][32];  // d x keys
    __shared__ __attribute__((aligned(16))) __bf16 Pl[4][16][32];

    const int tid  = threadIdx.x;
    const int lane = tid & 31;
    const int w    = tid >> 5;
    const int bh   = blockIdx.x >> 5;          // 32 q-groups per (b,h)
    const int qt   = (blockIdx.x & 31) * 4 + w;
    const int b    = bh >> 4;
    const int h    = bh & (HH - 1);

    // stage unit ids: 256 x 16B units per tile, 2 per thread
    const int u0 = tid, u1 = tid + 128;

    auto issue = [&](int jj, int nbuf) {
        const __bf16* kg0 = Kb  + ((size_t)bh * SS + jj * 32 + (u0 >> 3)) * HDD + (u0 & 7) * 8;
        const __bf16* kg1 = Kb  + ((size_t)bh * SS + jj * 32 + (u1 >> 3)) * HDD + (u1 & 7) * 8;
        const __bf16* vg0 = VTb + ((size_t)bh * HDD + (u0 >> 2)) * SS + jj * 32 + (u0 & 3) * 8;
        const __bf16* vg1 = VTb + ((size_t)bh * HDD + (u1 >> 2)) * SS + jj * 32 + (u1 & 3) * 8;
        async_b128(lds_off(&Kt[nbuf][0][0] + u0 * 8), kg0);
        async_b128(lds_off(&Kt[nbuf][0][0] + u1 * 8), kg1);
        async_b128(lds_off(&Vt[nbuf][0][0] + u0 * 8), vg0);
        async_b128(lds_off(&Vt[nbuf][0][0] + u1 * 8), vg1);
    };

    // Q fragments (16 x 64 = two 32-k chunks), held in registers all pass
    const __bf16* Qrow = Qb + ((size_t)bh * SS + qt * 16 + (lane & 15)) * HDD;
    const v16bf qf0 = load_frag(Qrow, 0,  lane);
    const v16bf qf1 = load_frag(Qrow, 32, lane);

    float rm[8], rl[8];
    v8f acc[4];
#pragma unroll
    for (int r = 0; r < 8; ++r) { rm[r] = -INFINITY; rl[r] = 0.0f; }
#pragma unroll
    for (int c = 0; c < 4; ++c) acc[c] = (v8f){};

    const int qm_base = qt * 16 + ((lane >> 4) << 3);
    const int ncol    = lane & 15;

    issue(0, 0);

    for (int j = 0; j < SS / 32; ++j) {
        const int cur = j & 1;
        if (j + 1 < SS / 32) { issue(j + 1, cur ^ 1); wait_async<4>(); }
        else                 { wait_async<0>(); }
        __syncthreads();   // all waves' staged K/V visible

        // preload both K-tile fragment pairs, then run the 4 score WMMAs
        v16bf kf[2][2];
#pragma unroll
        for (int t = 0; t < 2; ++t) {
            kf[t][0] = load_frag(&Kt[cur][t * 16 + ncol][0], 0,  lane);
            kf[t][1] = load_frag(&Kt[cur][t * 16 + ncol][0], 32, lane);
        }
        v8f s[2];
#pragma unroll
        for (int t = 0; t < 2; ++t) {
            v8f sc = {};
            sc = __builtin_amdgcn_wmma_f32_16x16x32_bf16(false, qf0, false, kf[t][0], (short)0, sc, false, false);
            sc = __builtin_amdgcn_wmma_f32_16x16x32_bf16(false, qf1, false, kf[t][1], (short)0, sc, false, false);

            const int kb0 = j * 32 + t * 16;
            const unsigned char mk = maskp[(size_t)b * SS + kb0 + ncol];
            const float* brow = bias + ((size_t)bh * SS + qm_base) * SS + kb0 + ncol;
#pragma unroll
            for (int r = 0; r < 8; ++r) {
                const float bv = brow[(size_t)r * SS];
                sc[r] += mk ? bv : NEGV;
            }
            s[t] = sc;
        }

        // online softmax, stage P (bf16) to per-wave LDS tile
#pragma unroll
        for (int r = 0; r < 8; ++r) {
            float rowm = redmax16(fmaxf(s[0][r], s[1][r]));
            const float mnew  = fmaxf(rm[r], rowm);
            const float alpha = __expf(rm[r] - mnew);
            const float p0 = __expf(s[0][r] - mnew);
            const float p1 = __expf(s[1][r] - mnew);
            rl[r] = rl[r] * alpha + redsum16(p0 + p1);
            rm[r] = mnew;
#pragma unroll
            for (int c = 0; c < 4; ++c) acc[c][r] *= alpha;
            const int mrow = ((lane >> 4) << 3) + r;
            Pl[w][mrow][ncol]      = f2bf(p0);
            Pl[w][mrow][16 + ncol] = f2bf(p1);
        }
        __builtin_amdgcn_wave_barrier();
        const v16bf pf = load_frag(&Pl[w][ncol][0], 0, lane);
        __builtin_amdgcn_wave_barrier();

        // O += P[16x32] * V[32x64] from staged V^T rows (preload all 4 frags)
        v16bf vf[4];
#pragma unroll
        for (int c = 0; c < 4; ++c)
            vf[c] = load_frag(&Vt[cur][c * 16 + ncol][0], 0, lane);
#pragma unroll
        for (int c = 0; c < 4; ++c)
            acc[c] = __builtin_amdgcn_wmma_f32_16x16x32_bf16(
                false, pf, false, vf[c], (short)0, acc[c], false, false);

        __syncthreads();   // done reading cur before it is overwritten
    }

    // normalize, store attn output bf16 [B, S, H*HD]
#pragma unroll
    for (int r = 0; r < 8; ++r) {
        const float inv = 1.0f / rl[r];
        const int q = qm_base + r;
        const size_t orow = ((size_t)b * SS + q) * DD + h * HDD;
#pragma unroll
        for (int c = 0; c < 4; ++c)
            attnb[orow + c * 16 + ncol] = f2bf(acc[c][r] * inv);
    }
}

// ---------------------------------------------------------------------------
// Host-side launcher
// ---------------------------------------------------------------------------
extern "C" void kernel_launch(void* const* d_in, const int* in_sizes, int n_in,
                              void* d_out, int out_size, void* d_ws, size_t ws_size,
                              hipStream_t stream) {
    const float*         x    = (const float*)d_in[0];
    const float*         pb   = (const float*)d_in[1];
    const unsigned char* mask = (const unsigned char*)d_in[2];
    const float*         Wqkv = (const float*)d_in[3];
    const float*         Wo   = (const float*)d_in[4];
    float*               out  = (float*)d_out;

    char* p = (char*)d_ws;
    auto alloc = [&](size_t bytes) -> void* {
        void* r = (void*)p;
        p += (bytes + 255) & ~(size_t)255;
        return r;
    };
    __bf16* xb    = (__bf16*)alloc((size_t)MM * DD * 2);
    __bf16* wqkvb = (__bf16*)alloc((size_t)NQKV * DD * 2);
    __bf16* wob   = (__bf16*)alloc((size_t)DD * DD * 2);
    __bf16* Qb    = (__bf16*)alloc((size_t)BB * HH * SS * HDD * 2);
    __bf16* Kb    = (__bf16*)alloc((size_t)BB * HH * SS * HDD * 2);
    __bf16* VTb   = (__bf16*)alloc((size_t)BB * HH * HDD * SS * 2);
    __bf16* attnb = (__bf16*)alloc((size_t)MM * DD * 2);

    // 1) fp32 -> bf16 conversions (vectorized x4)
    {
        int n4 = (MM * DD) / 4;
        cvt_f32_bf16<<<(n4 + 255) / 256, 256, 0, stream>>>(x, xb, n4);
        n4 = (NQKV * DD) / 4;
        cvt_f32_bf16<<<(n4 + 255) / 256, 256, 0, stream>>>(Wqkv, wqkvb, n4);
        n4 = (DD * DD) / 4;
        cvt_f32_bf16<<<(n4 + 255) / 256, 256, 0, stream>>>(Wo, wob, n4);
    }

    // 2) QKV projection: [4096,1024] x [3072,1024]^T, scatter to Q/K/V^T
    gemm_bf16_wmma<<<dim3(NQKV / 512, MM / 16), 256, 0, stream>>>(
        xb, wqkvb, MM, NQKV, DD, nullptr, Qb, Kb, VTb, 1);

    // 3) flash attention: 4 q-tiles (one (b,h)) per block, K/V LDS-shared
    attn_wmma<<<BB * HH * 32, 128, 0, stream>>>(
        Qb, Kb, VTb, pb, mask, attnb);

    // 4) output projection: [4096,1024] x [1024,1024]^T -> fp32 out
    gemm_bf16_wmma<<<dim3(DD / 512, MM / 16), 256, 0, stream>>>(
        attnb, wob, MM, DD, DD, out, nullptr, nullptr, nullptr, 0);
}
